// HexMeshQNet_50345606644282
// MI455X (gfx1250) — matrix operand order repaired and checked by reading, not compiled, not run on hardware
//
#include <hip/hip_runtime.h>
#include <math.h>

// ---------------------------------------------------------------------------
// HexMeshQNet for MI455X (gfx1250, wave32).
//  * Dense GEMMs run on v_wmma_f32_16x16x32_f16. LDS is laid out so every
//    per-lane fragment is a contiguous, 16B-aligned run: A row-major
//    (As[m][k]) and B transposed (Bs[n][k]) -> fragments load as
//    ds_load_b128 pairs instead of 16x ds_load_u16.
//  * Edge aggregation (the L2-atomic-bound part) uses float4 gathers
//    (global_load_b128) + f32 atomics; h (12.8 MB) lives in the 192 MB L2.
// ---------------------------------------------------------------------------

typedef __attribute__((ext_vector_type(16))) _Float16 v16h;
typedef __attribute__((ext_vector_type(8)))  _Float16 v8h;
typedef __attribute__((ext_vector_type(8)))  float    v8f;

#define FEATN 10
#define HIDN  64
#define SHEETS 256
#define SHEET_LEN 128

// ------------------------------- utility -----------------------------------

__global__ void k_zero(float* __restrict__ p, int n) {
  int i = blockIdx.x * blockDim.x + threadIdx.x;
  if (i < n) p[i] = 0.0f;
}

__global__ void k_degree(const int* __restrict__ dst, float* __restrict__ deg, int E) {
  int e = blockIdx.x * blockDim.x + threadIdx.x;
  if (e < E) atomicAdd(&deg[dst[e]], 1.0f);
}

__global__ void k_dinv(float* __restrict__ d, int n) {
  int i = blockIdx.x * blockDim.x + threadIdx.x;
  if (i < n) d[i] = rsqrtf(d[i] + 1.0f);  // +1 self loop
}

// ----------------------- generic WMMA GEMM (Ncols = 64) --------------------
// C[M x 64] = op(A[M x K] @ B[K x 64] + bias), op = ReLU if relu!=0.
// Block: 256 threads = 8 waves covering 32 rows x 64 cols.
// Wave w: row tile (w>>2)*16, col tile (w&3)*16.
// LDS strides padded to multiples of 16B (272B / 288B) for bank spread
// while keeping every fragment run 16B aligned (-> ds_load_b128).
__global__ void __launch_bounds__(256)
k_gemm_wmma(const float* __restrict__ A, int M, int K,
            const float* __restrict__ B, const float* __restrict__ bias,
            float* __restrict__ C, int relu) {
  __shared__ alignas(16) _Float16 As[32][136];   // 32 rows x Kpad(<=128), pad 8
  __shared__ alignas(16) _Float16 Bs[64][144];   // 64 cols x Kpad(<=128), pad 16

  const int tid  = threadIdx.x;
  const int Kpad = (K + 31) & ~31;               // 32 / 64 / 128
  const int rowBase = blockIdx.x * 32;

  // Stage B transposed (zero-padded in K): Bs[n][k] = B[k*64+n].
  for (int idx = tid; idx < Kpad * 64; idx += 256) {
    int k = idx >> 6, n = idx & 63;
    Bs[n][k] = (_Float16)((k < K) ? B[k * 64 + n] : 0.0f);
  }
  // Stage A row-major (zero-padded in rows and K).
  {
    int m = tid >> 3;                            // 8 threads per row, 32 rows
    int r = rowBase + m;
    for (int kk = (tid & 7); kk < Kpad; kk += 8) {
      As[m][kk] = (_Float16)((r < M && kk < K) ? A[(size_t)r * K + kk] : 0.0f);
    }
  }
  __syncthreads();

  const int wave = tid >> 5;                     // 0..7
  const int lane = tid & 31;
  const int half = lane >> 4;                    // lanes 0-15 vs 16-31
  const int l15  = lane & 15;
  const int rowTile = (wave >> 2) * 16;          // 0 or 16
  const int colBase = (wave & 3) * 16;           // 0/16/32/48

  const _Float16* arow = &As[rowTile + l15][0];
  const _Float16* brow = &Bs[colBase + l15][0];

  v8f c = {};
  for (int kc = 0; kc < Kpad; kc += 32) {
    // A 16-bit 16x32 layout: lane half 0 -> K {0..7,16..23}, half 1 -> +8.
    v8h a0 = *(const v8h*)(arow + kc + half * 8);
    v8h a1 = *(const v8h*)(arow + kc + 16 + half * 8);
    // B 16-bit 32x16 layout: lane = column; half 0 -> K {0..7,16..23}, half 1 -> +8.
    v8h b0 = *(const v8h*)(brow + kc + half * 8);
    v8h b1 = *(const v8h*)(brow + kc + 16 + half * 8);
    v16h a = __builtin_shufflevector(a0, a1, 0,1,2,3,4,5,6,7,8,9,10,11,12,13,14,15);
    v16h b = __builtin_shufflevector(b0, b1, 0,1,2,3,4,5,6,7,8,9,10,11,12,13,14,15);
    c = __builtin_amdgcn_wmma_f32_16x16x32_f16(
        /*neg_a=*/false, a, /*neg_b=*/false, b,
        /*c_mod=*/(short)0, c, /*reuse_a=*/false, /*reuse_b=*/false);
  }

  // f32 C/D layout: VGPR i -> row (i + 8*half), col = lane%16.
#pragma unroll
  for (int i = 0; i < 8; ++i) {
    int row = rowBase + rowTile + i + 8 * half;
    int col = colBase + l15;
    if (row < M) {
      float v = c[i] + (bias ? bias[col] : 0.0f);
      if (relu) v = fmaxf(v, 0.0f);
      C[(size_t)row * HIDN + col] = v;
    }
  }
}

// ------------------------- edge message passing -----------------------------
// agg[dst] += hW[src] * dinv[src]*dinv[dst]; 4 feats (float4) per thread.
__global__ void k_aggregate(const int* __restrict__ src, const int* __restrict__ dst,
                            const float* __restrict__ dinv, const float* __restrict__ hW,
                            float* __restrict__ agg, int E) {
  int t = blockIdx.x * blockDim.x + threadIdx.x;
  int e = t >> 4;
  if (e >= E) return;
  int f0 = (t & 15) << 2;
  int s = src[e], d = dst[e];
  float w = dinv[s] * dinv[d];
  const float4 hv = *(const float4*)(hW + (size_t)s * HIDN + f0);
  if (f0 == 0) {  // hint a future gather row into cache (global_prefetch_b8)
    int ep = e + 2048;
    if (ep < E) __builtin_prefetch(hW + (size_t)src[ep] * HIDN, 0, 1);
  }
  float* o = agg + (size_t)d * HIDN + f0;
  atomicAdd(o + 0, hv.x * w);
  atomicAdd(o + 1, hv.y * w);
  atomicAdd(o + 2, hv.z * w);
  atomicAdd(o + 3, hv.w * w);
}

// h = relu(agg + hW * dinv^2 + bias), in place on agg.
__global__ void k_finish(const float* __restrict__ hW, const float* __restrict__ dinv,
                         const float* __restrict__ bias, float* __restrict__ h_io, int N) {
  int i = blockIdx.x * blockDim.x + threadIdx.x;
  if (i >= N * HIDN) return;
  int n = i >> 6, f = i & 63;
  float di = dinv[n];
  float v = h_io[i] + hW[i] * di * di + bias[f];
  h_io[i] = fmaxf(v, 0.0f);
}

// -------------------------------- pooling -----------------------------------

__global__ void k_gsum(const float* __restrict__ h, float* __restrict__ gsum, int N) {
  int f = threadIdx.x;                       // 64 threads
  float acc = 0.0f;
  for (int r = blockIdx.x; r < N; r += gridDim.x) acc += h[(size_t)r * HIDN + f];
  atomicAdd(&gsum[f], acc);
}

__global__ void k_sheet_pool(const float* __restrict__ h, const int* __restrict__ idx,
                             float* __restrict__ sheets) {
  __shared__ int sidx[SHEET_LEN];
  int s = blockIdx.x, f = threadIdx.x;       // 64 threads
  for (int j = f; j < SHEET_LEN; j += 64) sidx[j] = idx[s * SHEET_LEN + j];
  __syncthreads();
  float acc = 0.0f;
  for (int j = 0; j < SHEET_LEN; ++j) acc += h[(size_t)sidx[j] * HIDN + f];
  sheets[s * HIDN + f] = acc * (1.0f / SHEET_LEN);
}

// ------------------------------- concat heads -------------------------------

__global__ void k_concat2(const float* __restrict__ A, const float* __restrict__ B,
                          float* __restrict__ out, int M) {
  int i = blockIdx.x * blockDim.x + threadIdx.x;
  if (i >= M * 128) return;
  int m = i >> 7, j = i & 127;
  out[i] = (j < 64) ? A[m * 64 + j] : B[m * 64 + (j - 64)];
}

__global__ void k_concat_bcast(const float* __restrict__ A, const float* __restrict__ g,
                               float scale, float* __restrict__ out, int M) {
  int i = blockIdx.x * blockDim.x + threadIdx.x;
  if (i >= M * 128) return;
  int m = i >> 7, j = i & 127;
  out[i] = (j < 64) ? A[m * 64 + j] : g[j - 64] * scale;
}

// q = hq @ q_W2 + q_b2  (64 -> 1)
__global__ void k_q2(const float* __restrict__ hq, const float* __restrict__ w,
                     const float* __restrict__ b, float* __restrict__ out) {
  int s = threadIdx.x;                       // 256 threads, one per sheet
  float acc = b[0];
#pragma unroll
  for (int k = 0; k < HIDN; ++k) acc += hq[s * HIDN + k] * w[k];
  out[s] = acc;
}

// ------------------------------- launcher -----------------------------------

extern "C" void kernel_launch(void* const* d_in, const int* in_sizes, int n_in,
                              void* d_out, int out_size, void* d_ws, size_t ws_size,
                              hipStream_t stream) {
  (void)n_in; (void)ws_size;
  const float* x        = (const float*)d_in[0];
  const int*   edge     = (const int*)  d_in[1];
  /* batch (d_in[2]) unused: single graph */
  const int*   sheetIdx = (const int*)  d_in[3];
  const float* sheetF   = (const float*)d_in[4];
  const float* W1 = (const float*)d_in[5],  *b1 = (const float*)d_in[6];
  const float* W2 = (const float*)d_in[7],  *b2 = (const float*)d_in[8];
  const float* W3 = (const float*)d_in[9],  *b3 = (const float*)d_in[10];
  const float* gW1 = (const float*)d_in[11], *gb1 = (const float*)d_in[12];
  const float* gW2 = (const float*)d_in[13], *gb2 = (const float*)d_in[14];
  const float* fW  = (const float*)d_in[15], *fb  = (const float*)d_in[16];
  const float* qW1 = (const float*)d_in[17], *qb1 = (const float*)d_in[18];
  const float* qW2 = (const float*)d_in[19], *qb2 = (const float*)d_in[20];
  float* out = (float*)d_out; (void)out_size;

  const int N = in_sizes[0] / FEATN;         // 50000
  const int E = in_sizes[1] / 2;             // 800000
  const int* src = edge;
  const int* dst = edge + E;

  // Workspace layout (floats).
  float* ws   = (float*)d_ws;
  float* dinv = ws;                                  size_t o = 50176;
  float* bufA = ws + o;  o += (size_t)N * HIDN;      // pre-agg h@W
  float* bufB = ws + o;  o += (size_t)N * HIDN;      // act (odd layers)
  float* bufC = ws + o;  o += (size_t)N * HIDN;      // act (even layers)
  float* gsum   = ws + o; o += 64;
  float* sheets = ws + o; o += SHEETS * HIDN;
  float* geoh   = ws + o; o += SHEETS * HIDN;
  float* geo    = ws + o; o += SHEETS * HIDN;
  float* cat    = ws + o; o += SHEETS * 128;
  float* fused  = ws + o; o += SHEETS * HIDN;
  float* hq     = ws + o; o += SHEETS * HIDN;

  const int NH = N * HIDN;
  dim3 blk(256);
  const int gN   = (N + 255) / 256;
  const int gNH  = (NH + 255) / 256;
  const int gE   = (E + 255) / 256;
  const int gAgg = (E * 16 + 255) / 256;
  const int gGemmN = (N + 31) / 32;
  const int gGemmS = (SHEETS + 31) / 32;

  // ---- degrees / symmetric norm ----
  k_zero<<<gN, blk, 0, stream>>>(dinv, N);
  k_degree<<<gE, blk, 0, stream>>>(dst, dinv, E);
  k_dinv<<<gN, blk, 0, stream>>>(dinv, N);

  // ---- GCN layer 1 (FEAT=10 -> 64) ----
  k_gemm_wmma<<<gGemmN, 256, 0, stream>>>(x, N, FEATN, W1, nullptr, bufA, 0);
  k_zero<<<gNH, blk, 0, stream>>>(bufB, NH);
  k_aggregate<<<gAgg, blk, 0, stream>>>(src, dst, dinv, bufA, bufB, E);
  k_finish<<<gNH, blk, 0, stream>>>(bufA, dinv, b1, bufB, N);

  // ---- GCN layer 2 (64 -> 64) ----
  k_gemm_wmma<<<gGemmN, 256, 0, stream>>>(bufB, N, HIDN, W2, nullptr, bufA, 0);
  k_zero<<<gNH, blk, 0, stream>>>(bufC, NH);
  k_aggregate<<<gAgg, blk, 0, stream>>>(src, dst, dinv, bufA, bufC, E);
  k_finish<<<gNH, blk, 0, stream>>>(bufA, dinv, b2, bufC, N);

  // ---- GCN layer 3 (64 -> 64) ----
  k_gemm_wmma<<<gGemmN, 256, 0, stream>>>(bufC, N, HIDN, W3, nullptr, bufA, 0);
  k_zero<<<gNH, blk, 0, stream>>>(bufB, NH);
  k_aggregate<<<gAgg, blk, 0, stream>>>(src, dst, dinv, bufA, bufB, E);
  k_finish<<<gNH, blk, 0, stream>>>(bufA, dinv, b3, bufB, N);   // h final: bufB

  // ---- pooling ----
  k_zero<<<1, 64, 0, stream>>>(gsum, 64);
  k_gsum<<<256, 64, 0, stream>>>(bufB, gsum, N);
  k_sheet_pool<<<SHEETS, 64, 0, stream>>>(bufB, sheetIdx, sheets);

  // ---- geo MLP: relu(sf@gW1+gb1)@gW2+gb2 ----
  k_gemm_wmma<<<gGemmS, 256, 0, stream>>>(sheetF, SHEETS, FEATN, gW1, gb1, geoh, 1);
  k_gemm_wmma<<<gGemmS, 256, 0, stream>>>(geoh, SHEETS, HIDN, gW2, gb2, geo, 0);

  // ---- fusion: relu([sheets|geo] @ fW + fb) ----
  k_concat2<<<(SHEETS * 128 + 255) / 256, blk, 0, stream>>>(sheets, geo, cat, SHEETS);
  k_gemm_wmma<<<gGemmS, 256, 0, stream>>>(cat, SHEETS, 128, fW, fb, fused, 1);

  // ---- q head: relu([fused|g_mean] @ qW1 + qb1) @ qW2 + qb2 ----
  k_concat_bcast<<<(SHEETS * 128 + 255) / 256, blk, 0, stream>>>(
      fused, gsum, 1.0f / (float)N, cat, SHEETS);
  k_gemm_wmma<<<gGemmS, 256, 0, stream>>>(cat, SHEETS, 128, qW1, qb1, hq, 1);
  k_q2<<<1, SHEETS, 0, stream>>>(hq, qW2, qb2, out);
}
